// PinSAGE_69346541961480
// MI455X (gfx1250) — matrix operand order repaired and checked by reading, not compiled, not run on hardware
//
#include <hip/hip_runtime.h>

// ---------------- types ----------------
typedef __attribute__((ext_vector_type(16))) _Float16 h16x16;
typedef __attribute__((ext_vector_type(8)))  _Float16 h16x8;
typedef __attribute__((ext_vector_type(4)))  _Float16 h16x4;
typedef __attribute__((ext_vector_type(8)))  float    f32x8;

// ---------------- problem constants ----------------
#define T_NB    20
#define D_IN    128
#define U_HID   128
#define E_OUT   128

#define ROWP  136   // padded f16 row length for gathered rows in LDS (272B)
#define ZROWP 264   // padded f16 row length for concat z in LDS (528B)

// ---------------- workspace layout (bytes) ----------------
#define WS_SUMSQ 0
#define WS_INV   64
#define WS_WQT   1024
#define WS_WNT   (WS_WQT + D_IN*U_HID*2)
#define WS_WWT   (WS_WNT + D_IN*U_HID*2)
#define WS_WGT   (WS_WWT + 2*U_HID*U_HID*2)
#define WS_EMB   (WS_WGT + U_HID*E_OUT*2)

// ---------------- WMMA helpers ----------------
static __device__ inline f32x8 wmma_f16(h16x16 a, h16x16 b, f32x8 c) {
  return __builtin_amdgcn_wmma_f32_16x16x32_f16(false, a, false, b, (short)0, c,
                                                false, false);
}

static __device__ inline h16x16 load_frag_pair(const _Float16* p0, const _Float16* p1) {
  h16x8 lo = *(const h16x8*)p0;
  h16x8 hi = *(const h16x8*)p1;
  return __builtin_shufflevector(lo, hi, 0,1,2,3,4,5,6,7,8,9,10,11,12,13,14,15);
}

// A fragment (16x32 f16): lane row m=lane%16, half=lane/16.
// VGPR0-3 hold K = half*8 + 0..7 ; VGPR4-7 hold K = 16 + half*8 + 0..7.
static __device__ inline h16x16 load_afrag(const _Float16* row_base, int ks, int half) {
  const int k0 = ks * 32 + half * 8;
  return load_frag_pair(row_base + k0, row_base + k0 + 16);
}

// B fragment (32x16 f16) from W^T stored row-major [U][K]:
// lane col n=lane%16 -> row (u) of W^T; K = ks*32 + half*8 + 0..7, then +16.
static __device__ inline h16x16 load_bfrag(const _Float16* wT, int u, int kstride,
                                           int ks, int half) {
  const _Float16* p = wT + (size_t)u * kstride + ks * 32 + half * 8;
  return load_frag_pair(p, p + 16);
}

// ---------------- utility kernels ----------------
__global__ void pinsage_init(float* sumsq) {
  if (threadIdx.x == 0 && blockIdx.x == 0) *sumsq = 0.f;
}

__global__ void pinsage_finish_norm(const float* sumsq, float* inv) {
  if (threadIdx.x == 0 && blockIdx.x == 0)
    *inv = rsqrtf(fmaxf(*sumsq, 1e-30f));
}

// dst[u][k] = (f16) src[k][u]   (src: K x U fp32 row-major)
__global__ void pinsage_transpose_f16(const float* __restrict__ src,
                                      _Float16* __restrict__ dst, int K, int U) {
  const int u = blockIdx.x;
  for (int k = threadIdx.x; k < K; k += blockDim.x)
    dst[(size_t)u * K + k] = (_Float16)src[(size_t)k * U + u];
}

// ---------------- pass 1: gather + node/neigh GEMMs + concat GEMM ----------------
__global__ __launch_bounds__(256) void pinsage_pass1(
    const int* __restrict__ node_ids, const int* __restrict__ neigh_ids,
    const float* __restrict__ alpha, const float* __restrict__ table,
    const float* __restrict__ bq, const float* __restrict__ bn,
    const float* __restrict__ bw,
    const _Float16* __restrict__ WqT, const _Float16* __restrict__ WnT,
    const _Float16* __restrict__ WwT,
    _Float16* __restrict__ emb_out, float* __restrict__ sumsq)
{
  __shared__ _Float16 rows_lds[(16 + 16 * T_NB) * ROWP];   // 91,392 B
  __shared__ _Float16 z_lds[16 * ZROWP];                   //  8,448 B
  __shared__ float    alpha_lds[16 * T_NB];                //  1,280 B

  const int tid  = threadIdx.x;
  const int lane = tid & 31;
  const int wave = tid >> 5;
  const int half = lane >> 4;
  const int nn   = lane & 15;
  const int n0   = blockIdx.x * 16;

  // stage alpha tile (rows n0..n0+15 are contiguous: 320 floats)
  if (tid < 16 * T_NB)
    alpha_lds[tid] = alpha[(size_t)n0 * T_NB + tid];

  // gather 16 node rows + 320 neighbor rows -> LDS (f16). One wave per row,
  // each lane moves a float4 (16B) and stores 4 f16 (8B).
  for (int r = wave; r < 16 + 16 * T_NB; r += 8) {
    int idx;
    if (r < 16) {
      idx = node_ids[n0 + r];
    } else {
      const int rr = r - 16;                 // rr = t*16 + nl
      idx = neigh_ids[(size_t)(n0 + (rr & 15)) * T_NB + (rr >> 4)];
    }
    float4 v = ((const float4*)(table + (size_t)idx * D_IN))[lane];
    h16x4 o;
    o[0] = (_Float16)v.x; o[1] = (_Float16)v.y;
    o[2] = (_Float16)v.z; o[3] = (_Float16)v.w;
    *(h16x4*)(rows_lds + r * ROWP + lane * 4) = o;
  }
  __syncthreads();

  const int uc  = wave * 16;     // this wave's output-column tile
  const int col = uc + nn;

  // ---- node branch: node_h = relu(node_x @ Wq + bq), cols [uc,uc+16) ----
  h16x16 w0 = load_bfrag(WqT, col, D_IN, 0, half);
  h16x16 w1 = load_bfrag(WqT, col, D_IN, 1, half);
  h16x16 w2 = load_bfrag(WqT, col, D_IN, 2, half);
  h16x16 w3 = load_bfrag(WqT, col, D_IN, 3, half);
  {
    const _Float16* nrow = rows_lds + nn * ROWP;
    f32x8 acc = {0.f,0.f,0.f,0.f,0.f,0.f,0.f,0.f};
    acc = wmma_f16(load_afrag(nrow, 0, half), w0, acc);
    acc = wmma_f16(load_afrag(nrow, 1, half), w1, acc);
    acc = wmma_f16(load_afrag(nrow, 2, half), w2, acc);
    acc = wmma_f16(load_afrag(nrow, 3, half), w3, acc);
    const float bias = bq[col];
#pragma unroll
    for (int r = 0; r < 8; ++r) {
      float h = fmaxf(acc[r] + bias, 0.f);
      z_lds[(r + 8 * half) * ZROWP + col] = (_Float16)h;
    }
  }

  // ---- neighbor branch: agg = sum_t alpha[n,t]*relu(x_t @ Wn + bn) ----
  w0 = load_bfrag(WnT, col, D_IN, 0, half);
  w1 = load_bfrag(WnT, col, D_IN, 1, half);
  w2 = load_bfrag(WnT, col, D_IN, 2, half);
  w3 = load_bfrag(WnT, col, D_IN, 3, half);
  const float biasn = bn[col];
  f32x8 agg = {0.f,0.f,0.f,0.f,0.f,0.f,0.f,0.f};
  for (int t = 0; t < T_NB; ++t) {
    const _Float16* arow = rows_lds + (16 + t * 16 + nn) * ROWP;
    f32x8 d = {0.f,0.f,0.f,0.f,0.f,0.f,0.f,0.f};
    d = wmma_f16(load_afrag(arow, 0, half), w0, d);
    d = wmma_f16(load_afrag(arow, 1, half), w1, d);
    d = wmma_f16(load_afrag(arow, 2, half), w2, d);
    d = wmma_f16(load_afrag(arow, 3, half), w3, d);
#pragma unroll
    for (int r = 0; r < 8; ++r) {
      float h = fmaxf(d[r] + biasn, 0.f);
      agg[r] += alpha_lds[(r + 8 * half) * T_NB + t] * h;
    }
  }
#pragma unroll
  for (int r = 0; r < 8; ++r)
    z_lds[(r + 8 * half) * ZROWP + U_HID + col] = (_Float16)agg[r];
  __syncthreads();

  // ---- emb_pre = relu([node_h | agg] @ Ww + bw), K = 256 ----
  float hv[8];
  float ss = 0.f;
  {
    const _Float16* zrow = z_lds + nn * ZROWP;
    f32x8 e = {0.f,0.f,0.f,0.f,0.f,0.f,0.f,0.f};
#pragma unroll
    for (int ks = 0; ks < 8; ++ks) {
      h16x16 a = load_afrag(zrow, ks, half);
      h16x16 b = load_bfrag(WwT, col, 2 * U_HID, ks, half);
      e = wmma_f16(a, b, e);
    }
    const float biasw = bw[col];
#pragma unroll
    for (int r = 0; r < 8; ++r) {
      hv[r] = fmaxf(e[r] + biasw, 0.f);
      ss += hv[r] * hv[r];
    }
  }
  // wave-level reduction of sum-of-squares, one atomic per wave
#pragma unroll
  for (int off = 16; off > 0; off >>= 1)
    ss += __shfl_xor(ss, off);
  if (lane == 0) atomicAdd(sumsq, ss);

  // repack the 16x128 emb tile through LDS (reuse z_lds) for one fully
  // coalesced 4KB burst store instead of scattered b16 stores.
  __syncthreads();                      // all waves done reading z_lds A-frags
#pragma unroll
  for (int r = 0; r < 8; ++r)
    z_lds[(r + 8 * half) * U_HID + col] = (_Float16)hv[r];
  __syncthreads();
  *((h16x8*)(emb_out + (size_t)n0 * U_HID) + tid) =
      *(const h16x8*)(z_lds + tid * 8);
}

// ---------------- pass 2: out = relu((emb_pre @ Wg) * inv_norm + bg) ----------------
__global__ __launch_bounds__(256) void pinsage_pass2(
    const _Float16* __restrict__ emb, const _Float16* __restrict__ WgT,
    const float* __restrict__ bg, const float* __restrict__ inv_norm,
    float* __restrict__ out)
{
  __shared__ float o_lds[16 * E_OUT];   // 8KB staging for coalesced stores

  const int tid  = threadIdx.x;
  const int lane = tid & 31;
  const int wave = tid >> 5;
  const int half = lane >> 4;
  const int nn   = lane & 15;
  const int n0   = blockIdx.x * 16;
  const int col  = wave * 16 + nn;

  const _Float16* arow = emb + (size_t)(n0 + nn) * U_HID;
  f32x8 e = {0.f,0.f,0.f,0.f,0.f,0.f,0.f,0.f};
#pragma unroll
  for (int ks = 0; ks < 4; ++ks) {
    h16x16 a = load_afrag(arow, ks, half);
    h16x16 b = load_bfrag(WgT, col, U_HID, ks, half);
    e = wmma_f16(a, b, e);
  }
  const float inv  = *inv_norm;
  const float bias = bg[col];
#pragma unroll
  for (int r = 0; r < 8; ++r)
    o_lds[(r + 8 * half) * E_OUT + col] = fmaxf(e[r] * inv + bias, 0.f);
  __syncthreads();
  // 16 rows x 128 f32 = 8KB, one contiguous 32B slice per thread
  float4* op = (float4*)(out + (size_t)n0 * E_OUT);
  const float4* lp = (const float4*)o_lds;
  op[tid * 2 + 0] = lp[tid * 2 + 0];
  op[tid * 2 + 1] = lp[tid * 2 + 1];
}

// ---------------- launcher ----------------
extern "C" void kernel_launch(void* const* d_in, const int* in_sizes, int n_in,
                              void* d_out, int out_size, void* d_ws, size_t ws_size,
                              hipStream_t stream) {
  const int*   node_ids  = (const int*)d_in[0];
  const int*   neigh_ids = (const int*)d_in[1];
  const float* alpha     = (const float*)d_in[2];
  const float* table     = (const float*)d_in[3];
  const float* Wq        = (const float*)d_in[4];
  const float* bq        = (const float*)d_in[5];
  const float* Wn        = (const float*)d_in[6];
  const float* bn        = (const float*)d_in[7];
  const float* Ww        = (const float*)d_in[8];
  const float* bw        = (const float*)d_in[9];
  const float* Wg        = (const float*)d_in[10];
  const float* bg        = (const float*)d_in[11];

  char* ws = (char*)d_ws;
  float*    sumsq = (float*)(ws + WS_SUMSQ);
  float*    invn  = (float*)(ws + WS_INV);
  _Float16* wqt   = (_Float16*)(ws + WS_WQT);
  _Float16* wnt   = (_Float16*)(ws + WS_WNT);
  _Float16* wwt   = (_Float16*)(ws + WS_WWT);
  _Float16* wgt   = (_Float16*)(ws + WS_WGT);
  _Float16* embp  = (_Float16*)(ws + WS_EMB);

  const int n = in_sizes[0];          // 50000, multiple of 16
  const int nblocks = n / 16;

  pinsage_init<<<1, 1, 0, stream>>>(sumsq);
  pinsage_transpose_f16<<<U_HID, 128, 0, stream>>>(Wq, wqt, D_IN, U_HID);
  pinsage_transpose_f16<<<U_HID, 128, 0, stream>>>(Wn, wnt, D_IN, U_HID);
  pinsage_transpose_f16<<<U_HID, 256, 0, stream>>>(Ww, wwt, 2 * U_HID, U_HID);
  pinsage_transpose_f16<<<E_OUT, 128, 0, stream>>>(Wg, wgt, U_HID, E_OUT);

  pinsage_pass1<<<nblocks, 256, 0, stream>>>(node_ids, neigh_ids, alpha, table,
                                             bq, bn, bw, wqt, wnt, wwt,
                                             embp, sumsq);
  pinsage_finish_norm<<<1, 1, 0, stream>>>(sumsq, invn);
  pinsage_pass2<<<nblocks, 256, 0, stream>>>(embp, wgt, bg, invn, (float*)d_out);
}
